// CrossAttentionT2S_38543036514771
// MI455X (gfx1250) — compile-verified, matrix operand order checked by reading, not run on hardware
//
#include <hip/hip_runtime.h>
#include <hip/hip_bf16.h>
#include <math.h>

// ---------------------------------------------------------------------------
// CrossAttentionT2S for MI455X (gfx1250, wave32, WMMA + async-to-LDS DMA).
//
// Pipeline (all on `stream`):
//   0) cast q_w / kv_w / proj_w fp32 -> f16               (weights are [N,K])
//   1) prep: t_x -> t16[b,n,t,c]  (+vmae_time_pos, f16)   rows = 8*196*8 = 12544
//            s_x -> s16[b,hw,t,c] (+cnn_time_pos,  f16)   rows = 8*784*8 = 50176
//   2) KV GEMM  (12544 x 768 x 1536) f16 WMMA, f32 acc, +kv_b   -> kv16 (f16)
//      *** computed per ViT token, 4x less work than per-pixel ***
//   3) Q  GEMM  (50176 x 768 x  768) +q_b, *0.125               -> q16  (f16)
//   4) attention: per (b, vit-token) block; K/V shared by 2x2 pixel quad,
//      staged in LDS; 8x8 softmax per (pixel,head) in VALU       -> o16  (f16)
//   5) proj GEMM (50176 x 768 x 768) +proj_b, scatter fp32 to (b t) h w c
//
// GEMM staging uses GLOBAL_LOAD_ASYNC_TO_LDS_B128 (ASYNCcnt) with a
// double-buffered software pipeline when the builtin is available; falls
// back to global_load_b128 + ds_store_b128 otherwise.
// ---------------------------------------------------------------------------

typedef __attribute__((ext_vector_type(16))) _Float16 v16h;
typedef __attribute__((ext_vector_type(8)))  _Float16 v8h;
typedef __attribute__((ext_vector_type(4)))  _Float16 v4h;
typedef __attribute__((ext_vector_type(8)))  float    v8f;
typedef __attribute__((ext_vector_type(4)))  int      v4i;

union Frag16 { v16h v; v8h h[2]; };

#define DIM   768
#define NHEAD 12
#define HD    64
#define TT    8
#define BB    8
#define HH    28
#define HW    784     // 28*28
#define NVIT  196     // 14*14
#define KV_N  1536

#if defined(__gfx1250__) &&                                              \
    __has_builtin(__builtin_amdgcn_global_load_async_to_lds_b128) &&     \
    __has_builtin(__builtin_amdgcn_s_wait_asynccnt)
#define USE_ASYNC_LDS 1
#else
#define USE_ASYNC_LDS 0
#endif

#if USE_ASYNC_LDS
// Probe-derived signature: (v4i addrspace(1)* gsrc, v4i addrspace(3)* ldst,
//                           imm int offset, imm int cpol); tracked by ASYNCcnt.
typedef __attribute__((address_space(1))) v4i* GP128;
typedef __attribute__((address_space(3))) v4i* SP128;
__device__ __forceinline__ void async_copy_b128(const void* g, void* s) {
  __builtin_amdgcn_global_load_async_to_lds_b128((GP128)(v4i*)g,
                                                 (SP128)(v4i*)s, 0, 0);
}
#endif

// ---------------------------------------------------------------------------
// Weight cast fp32 -> f16 (4 elements / thread)
// ---------------------------------------------------------------------------
__global__ __launch_bounds__(256) void cast_w_f16(const float* __restrict__ in,
                                                  _Float16* __restrict__ out,
                                                  long n4) {
  long gid = (long)blockIdx.x * blockDim.x + threadIdx.x;
  if (gid >= n4) return;
  float4 x = ((const float4*)in)[gid];
  v4h o;
  o[0] = (_Float16)x.x; o[1] = (_Float16)x.y;
  o[2] = (_Float16)x.z; o[3] = (_Float16)x.w;
  ((v4h*)out)[gid] = o;
}

// ---------------------------------------------------------------------------
// in  : [B, T, NIDX, C] fp32
// out : [B, NIDX, T, C] f16 = in + pos[t]   (time-major -> token-major)
// ---------------------------------------------------------------------------
__global__ __launch_bounds__(256) void prep_cast_pos(const float* __restrict__ in,
                                                     const float* __restrict__ pos,
                                                     _Float16* __restrict__ out,
                                                     int NIDX, long total4) {
  long gid = (long)blockIdx.x * blockDim.x + threadIdx.x;
  if (gid >= total4) return;
  long e = gid * 4;
  int  c  = (int)(e % DIM);
  long r  = e / DIM;
  int  tt = (int)(r % TT);
  long r2 = r / TT;
  int idx = (int)(r2 % NIDX);
  int b   = (int)(r2 / NIDX);
  float4 x = *(const float4*)(in + (((long)(b * TT + tt) * NIDX + idx) * DIM + c));
  float4 p = *(const float4*)(pos + (long)tt * DIM + c);
  v4h o;
  o[0] = (_Float16)(x.x + p.x); o[1] = (_Float16)(x.y + p.y);
  o[2] = (_Float16)(x.z + p.z); o[3] = (_Float16)(x.w + p.w);
  *(v4h*)(out + e) = o;
}

// ---------------------------------------------------------------------------
// WMMA GEMM: C[M,N] = A[M,768] * W[N,768]^T  (f16 in, f32 accumulate)
//   block tile 128x128, 8 waves (4x2), wave tile 32x64 = 2x4 WMMA 16x16x32
//   Double-buffered LDS tiles, padded to 40-halfword (80 B) row stride:
//   16 B aligned and bank-conflict-free fragment loads.
// Fragment layout per ISA 7.12.2 (16-bit A 16x32):
//   lanes 0-15  hold K {0..7,16..23}; lanes 16-31 hold K {8..15,24..31}
// Async pipeline: tile k+2 DMA issued before tile-k WMMAs; each tile is
// 4 async b128 ops per thread (in-order completion) -> s_wait_asynccnt 4.
// EPI 0: store f16 to Cout[row*Ncols+col], val=(acc+bias)*scale
// EPI 1: store fp32 scattered to (b t) h w c final layout, +bias
// ---------------------------------------------------------------------------
template <int EPI>
__global__ __launch_bounds__(256) void gemm_f16_wmma(
    const _Float16* __restrict__ A,    // [M, 768] row-major f16
    const _Float16* __restrict__ Wm,   // [N, 768] row-major f16 (B^T natural)
    const float*    __restrict__ bias, // [N]
    void* __restrict__ Cout,
    int Ncols, float outScale) {
  constexpr int K      = DIM;
  constexpr int KT     = 32;
  constexpr int NKT    = K / KT;  // 24
  constexpr int LDSROW = 40;      // halfwords per LDS tile row (32 data + 8 pad)

  __shared__ __align__(16) _Float16 lA[2][128 * LDSROW];
  __shared__ __align__(16) _Float16 lB[2][128 * LDSROW];

  const int  tid     = threadIdx.x;
  const int  lane    = tid & 31;           // wave32
  const int  wid     = tid >> 5;           // 8 waves
  const int  wm      = wid & 3;            // wave row   (0..3) -> 32 rows
  const int  wn      = wid >> 2;           // wave col   (0..1) -> 64 cols
  const long rowBase = (long)blockIdx.x * 128;
  const int  colBase = (int)blockIdx.y * 128;

  const int hseg = (lane >> 4) * 8;        // K sub-offset (0 or 8)
  const int rsel = lane & 15;              // M (or N) index within 16

  // Stage one 128x32 f16 tile of A and of W into LDS buffer `buf` for K-chunk
  // kc. 512 x 16B chunks per tile; 2 chunks of each matrix per thread.
  auto issue_tile = [&](int buf, int kc) {
    const int kO = kc * KT;
#pragma unroll
    for (int i = 0; i < 2; ++i) {
      const int c   = tid + i * 256;
      const int row = c >> 2;
      const int off = (c & 3) * 8;  // halfword offset within K-chunk
#if USE_ASYNC_LDS
      async_copy_b128(A + (rowBase + row) * K + kO + off,
                      &lA[buf][row * LDSROW + off]);
      async_copy_b128(Wm + (long)(colBase + row) * K + kO + off,
                      &lB[buf][row * LDSROW + off]);
#else
      *(v8h*)(&lA[buf][row * LDSROW + off]) =
          *(const v8h*)(A + (rowBase + row) * K + kO + off);
      *(v8h*)(&lB[buf][row * LDSROW + off]) =
          *(const v8h*)(Wm + (long)(colBase + row) * K + kO + off);
#endif
    }
#if !USE_ASYNC_LDS
    if (kc + 1 < NKT) {  // hint next K-tile toward GL2 (global_prefetch_b8)
      __builtin_prefetch(A + (rowBase + (tid >> 1)) * K + kO + KT, 0, 2);
      __builtin_prefetch(Wm + (long)(colBase + (tid >> 1)) * K + kO + KT, 0, 2);
    }
#endif
  };

  v8f acc[2][4];
#pragma unroll
  for (int mi = 0; mi < 2; ++mi)
#pragma unroll
    for (int ni = 0; ni < 4; ++ni) acc[mi][ni] = (v8f){0, 0, 0, 0, 0, 0, 0, 0};

  // Prologue: fill both buffers.
  issue_tile(0, 0);
  issue_tile(1, 1);

  for (int kc = 0; kc < NKT; ++kc) {
#if USE_ASYNC_LDS
    // Own async ops complete in order: allow the next tile's 4 to remain
    // outstanding while guaranteeing tile kc's 4 have landed in LDS.
    if (kc + 1 < NKT) __builtin_amdgcn_s_wait_asynccnt(4);
    else              __builtin_amdgcn_s_wait_asynccnt(0);
#endif
    __syncthreads();  // all waves see buffer `kc&1` filled

    const int buf = kc & 1;
    Frag16 a[2], b[4];
#pragma unroll
    for (int mi = 0; mi < 2; ++mi) {
      const int r = wm * 32 + mi * 16 + rsel;
      a[mi].h[0] = *(const v8h*)(&lA[buf][r * LDSROW + hseg]);
      a[mi].h[1] = *(const v8h*)(&lA[buf][r * LDSROW + 16 + hseg]);
    }
#pragma unroll
    for (int ni = 0; ni < 4; ++ni) {
      const int r = wn * 64 + ni * 16 + rsel;
      b[ni].h[0] = *(const v8h*)(&lB[buf][r * LDSROW + hseg]);
      b[ni].h[1] = *(const v8h*)(&lB[buf][r * LDSROW + 16 + hseg]);
    }
    __syncthreads();  // everyone done reading buffer `buf` (barrier waits DScnt)

    // Refill the just-consumed buffer for tile kc+2 -> DMA overlaps the WMMAs.
    if (kc + 2 < NKT) issue_tile(buf, kc + 2);

#pragma unroll
    for (int mi = 0; mi < 2; ++mi)
#pragma unroll
      for (int ni = 0; ni < 4; ++ni)
        acc[mi][ni] = __builtin_amdgcn_wmma_f32_16x16x32_f16(
            false, a[mi].v, false, b[ni].v, (short)0, acc[mi][ni], false, false);
  }

  // Epilogue. C/D layout: VGPR r -> M=r (lanes 0-15) / M=r+8 (lanes 16-31),
  // N = lane&15.
#pragma unroll
  for (int mi = 0; mi < 2; ++mi) {
#pragma unroll
    for (int ni = 0; ni < 4; ++ni) {
      const int   col = colBase + wn * 64 + ni * 16 + rsel;
      const float bb  = bias[col];
#pragma unroll
      for (int r8 = 0; r8 < 8; ++r8) {
        const long  row = rowBase + wm * 32 + mi * 16 + r8 + (lane >> 4) * 8;
        const float v   = (acc[mi][ni][r8] + bb) * outScale;
        if (EPI == 0) {
          ((_Float16*)Cout)[row * Ncols + col] = (_Float16)v;
        } else {
          // row = (b*784 + hw)*8 + t  ->  out[(b*8+t)*784 + hw][col] fp32
          const int  tq  = (int)(row & 7);
          const long rhw = row >> 3;
          const int  hw  = (int)(rhw % HW);
          const int  bq  = (int)(rhw / HW);
          ((float*)Cout)[((long)(bq * TT + tq) * HW + hw) * DIM + col] = v;
        }
      }
    }
  }
}

// ---------------------------------------------------------------------------
// Attention: one block per (b, ViT token n). The 2x2 pixel quad shares K/V,
// which is staged once in LDS (2 x 8 x 768 f16 = 24 KB). 384 threads =
// (4 pixels x 12 heads x 8 query rows); each thread computes one softmax row.
// FLOP count here is ~1% of the GEMMs -> VALU is the right tool.
// ---------------------------------------------------------------------------
__global__ __launch_bounds__(384) void attn_small(
    const _Float16* __restrict__ q16,   // [(b,hw,t), 768] pre-scaled by 0.125
    const _Float16* __restrict__ kv16,  // [(b,n,t), 1536] k=cols[0,768) v=[768,1536)
    _Float16* __restrict__ o16) {       // [(b,hw,t), 768]
  const int blk = blockIdx.x;           // b*196 + n
  const int b   = blk / NVIT;
  const int n   = blk % NVIT;
  const int th  = n / 14, tw = n % 14;

  __shared__ __align__(16) _Float16 kf[TT][DIM];
  __shared__ __align__(16) _Float16 vf[TT][DIM];

  const int tid = threadIdx.x;
  for (int i = tid; i < TT * KV_N / 8; i += 384) {
    const int e  = i * 8;
    const int tt = e / KV_N;
    const int c  = e % KV_N;
    v8h d = *(const v8h*)(kv16 + ((long)blk * TT + tt) * KV_N + c);
    if (c < DIM) *(v8h*)(&kf[tt][c]) = d;
    else         *(v8h*)(&vf[tt][c - DIM]) = d;
  }
  __syncthreads();

  const int pix = tid / 96;        // 0..3 within the 2x2 quad
  const int hd_ = (tid % 96) / 8;  // head 0..11
  const int tq  = tid % 8;         // query time index
  const int h   = th * 2 + (pix >> 1);
  const int w   = tw * 2 + (pix & 1);
  const long qrow = ((long)b * HW + h * HH + w) * TT + tq;

  const _Float16* qp = q16 + qrow * DIM + hd_ * HD;
  v8h qv[8];
#pragma unroll
  for (int i = 0; i < 8; ++i) qv[i] = *(const v8h*)(qp + i * 8);

  float logit[TT];
#pragma unroll
  for (int kt = 0; kt < TT; ++kt) {
    float s = 0.f;
    const _Float16* kp = &kf[kt][hd_ * HD];
#pragma unroll
    for (int i = 0; i < 8; ++i) {
      const v8h kk = *(const v8h*)(kp + i * 8);
#pragma unroll
      for (int j = 0; j < 8; ++j) s += (float)qv[i][j] * (float)kk[j];
    }
    logit[kt] = s;
  }

  float m = logit[0];
#pragma unroll
  for (int kt = 1; kt < TT; ++kt) m = fmaxf(m, logit[kt]);
  float p[TT], sum = 0.f;
#pragma unroll
  for (int kt = 0; kt < TT; ++kt) { p[kt] = __expf(logit[kt] - m); sum += p[kt]; }
  const float inv = 1.f / sum;

  float o[HD];
#pragma unroll
  for (int i = 0; i < HD; ++i) o[i] = 0.f;
#pragma unroll
  for (int kt = 0; kt < TT; ++kt) {
    const float pw = p[kt];
    const _Float16* vp = &vf[kt][hd_ * HD];
#pragma unroll
    for (int i = 0; i < 8; ++i) {
      const v8h vv = *(const v8h*)(vp + i * 8);
#pragma unroll
      for (int j = 0; j < 8; ++j) o[i * 8 + j] += pw * (float)vv[j];
    }
  }

  _Float16* op = o16 + qrow * DIM + hd_ * HD;
#pragma unroll
  for (int i = 0; i < 8; ++i) {
    v8h ov;
#pragma unroll
    for (int j = 0; j < 8; ++j) ov[j] = (_Float16)(o[i * 8 + j] * inv);
    *(v8h*)(op + i * 8) = ov;
  }
}

// ---------------------------------------------------------------------------
extern "C" void kernel_launch(void* const* d_in, const int* /*in_sizes*/, int /*n_in*/,
                              void* d_out, int /*out_size*/, void* d_ws, size_t /*ws_size*/,
                              hipStream_t stream) {
  const float* s_x  = (const float*)d_in[0];
  const float* t_x  = (const float*)d_in[1];
  const float* cpos = (const float*)d_in[2];
  const float* vpos = (const float*)d_in[3];
  const float* q_w  = (const float*)d_in[4];
  const float* q_b  = (const float*)d_in[5];
  const float* kv_w = (const float*)d_in[6];
  const float* kv_b = (const float*)d_in[7];
  const float* p_w  = (const float*)d_in[8];
  const float* p_b  = (const float*)d_in[9];
  float* out = (float*)d_out;

  // Workspace carve (~294 MB total; all f16 intermediates, 256 B aligned).
  char* ws = (char*)d_ws;
  auto carve = [&](size_t bytes) {
    void* p = (void*)ws;
    ws += (bytes + 255) & ~(size_t)255;
    return p;
  };
  const long R_KV = (long)BB * NVIT * TT;  // 12544
  const long R_Q  = (long)BB * HW * TT;    // 50176
  _Float16* qw16  = (_Float16*)carve((size_t)DIM * DIM * 2);
  _Float16* kvw16 = (_Float16*)carve((size_t)KV_N * DIM * 2);
  _Float16* pw16  = (_Float16*)carve((size_t)DIM * DIM * 2);
  _Float16* t16   = (_Float16*)carve((size_t)R_KV * DIM * 2);
  _Float16* s16   = (_Float16*)carve((size_t)R_Q * DIM * 2);
  _Float16* kv16  = (_Float16*)carve((size_t)R_KV * KV_N * 2);
  _Float16* q16   = (_Float16*)carve((size_t)R_Q * DIM * 2);
  _Float16* o16   = (_Float16*)carve((size_t)R_Q * DIM * 2);

  // 0) weights -> f16
  {
    long n4 = (long)DIM * DIM / 4;
    cast_w_f16<<<(int)((n4 + 255) / 256), 256, 0, stream>>>(q_w, qw16, n4);
    cast_w_f16<<<(int)((n4 + 255) / 256), 256, 0, stream>>>(p_w, pw16, n4);
    long n4kv = (long)KV_N * DIM / 4;
    cast_w_f16<<<(int)((n4kv + 255) / 256), 256, 0, stream>>>(kv_w, kvw16, n4kv);
  }
  // 1) activations: transpose to token-major, add time-pos, cast f16
  {
    long n4 = R_KV * DIM / 4;
    prep_cast_pos<<<(int)((n4 + 255) / 256), 256, 0, stream>>>(t_x, vpos, t16, NVIT, n4);
  }
  {
    long n4 = R_Q * DIM / 4;
    prep_cast_pos<<<(int)((n4 + 255) / 256), 256, 0, stream>>>(s_x, cpos, s16, HW, n4);
  }
  // 2) KV projection per ViT token (4x less work than per pixel)
  gemm_f16_wmma<0><<<dim3((unsigned)(R_KV / 128), KV_N / 128), 256, 0, stream>>>(
      t16, kvw16, kv_b, kv16, KV_N, 1.0f);
  // 3) Q projection (+bias, *head_dim^-0.5)
  gemm_f16_wmma<0><<<dim3((unsigned)(R_Q / 128), DIM / 128), 256, 0, stream>>>(
      s16, qw16, q_b, q16, DIM, 0.125f);
  // 4) attention (K/V shared per 2x2 quad, LDS-staged)
  attn_small<<<BB * NVIT, 384, 0, stream>>>(q16, kv16, o16);
  // 5) output projection + bias, scatter fp32 to (b t) h w c
  gemm_f16_wmma<1><<<dim3((unsigned)(R_Q / 128), DIM / 128), 256, 0, stream>>>(
      o16, pw16, p_b, out, DIM, 1.0f);
}
